// MultiHeadAttention_7791070675444
// MI455X (gfx1250) — matrix operand order repaired and checked by reading, not compiled, not run on hardware
//
#include <hip/hip_runtime.h>
#include <hip/hip_bf16.h>

// ---------------- types ----------------
typedef __attribute__((ext_vector_type(16))) __bf16 v16bf;
typedef __attribute__((ext_vector_type(8)))  __bf16 v8bf;
typedef __attribute__((ext_vector_type(8)))  float  v8f;
typedef __attribute__((ext_vector_type(4)))  float  v4f;

#define BATCH 2
#define SEQ   2048
#define EMB   1024
#define HEADS 16
#define HDIM  64
#define ROWS  (BATCH * SEQ)   // 4096

// ---------------- WMMA wrapper ----------------
__device__ __forceinline__ v8f wmma_bf16(v16bf a, v16bf b, v8f c) {
    // D = A(16x32) * B(32x16) + C, f32 accumulate
    return __builtin_amdgcn_wmma_f32_16x16x32_bf16(
        /*neg_a=*/false, a, /*neg_b=*/false, b,
        /*c_mod=*/(short)0, c, /*reuse_a=*/false, /*reuse_b=*/false);
}

// ---------------- operand loaders (ISA 7.12.2 layouts, wave32) ----------------
// A 16x32 bf16: lane l(0-15) row M=l, halves {K 0..7, K 16..23}; lanes 16-31
// same rows, halves {K 8..15, K 24..31}.  hf = lane>>4, l = lane&15.
__device__ __forceinline__ v16bf load_a_bf16(const __bf16* __restrict__ base,
                                             int lda, int mBase, int k0,
                                             int l, int hf) {
    const __bf16* p = base + (size_t)(mBase + l) * lda + k0 + hf * 8;
    v8bf c0 = *(const v8bf*)p;
    v8bf c1 = *(const v8bf*)(p + 16);
    v16bf a;
#pragma unroll
    for (int i = 0; i < 8; ++i) { a[i] = c0[i]; a[i + 8] = c1[i]; }
    return a;
}

__device__ __forceinline__ v16bf load_a_f32(const float* __restrict__ base,
                                            int lda, int mBase, int k0,
                                            int l, int hf) {
    const float* p = base + (size_t)(mBase + l) * lda + k0 + hf * 8;
    v4f c0 = *(const v4f*)p;
    v4f c1 = *(const v4f*)(p + 4);
    v4f c2 = *(const v4f*)(p + 16);
    v4f c3 = *(const v4f*)(p + 20);
    v16bf a;
#pragma unroll
    for (int i = 0; i < 4; ++i) {
        a[i]      = (__bf16)c0[i];
        a[i + 4]  = (__bf16)c1[i];
        a[i + 8]  = (__bf16)c2[i];
        a[i + 12] = (__bf16)c3[i];
    }
    return a;
}

// B 32x16 bf16 in NT form: B[k][n] = Src[n0+n][k0+k]  (contraction dim is the
// fast axis of Src).  Lane l holds column n=l, halves = K {hf*8..+7, 16+hf*8..+7}
// (same K order as the A loaders), two 16B reads per lane, zero conversion VALU.
__device__ __forceinline__ v16bf load_b_nt_bf16(const __bf16* __restrict__ src,
                                                int ldb, int n0, int k0,
                                                int l, int hf) {
    const __bf16* p = src + (size_t)(n0 + l) * ldb + k0 + hf * 8;
    v8bf c0 = *(const v8bf*)p;
    v8bf c1 = *(const v8bf*)(p + 16);
    v16bf bv;
#pragma unroll
    for (int i = 0; i < 8; ++i) { bv[i] = c0[i]; bv[i + 8] = c1[i]; }
    return bv;
}

// ---------------- one-time f32 -> bf16 weight conversion ---------------------
__global__ void __launch_bounds__(256)
cvt_f32_bf16_kernel(const float* __restrict__ src, __bf16* __restrict__ dst) {
    const size_t i = ((size_t)blockIdx.x * blockDim.x + threadIdx.x) * 8;
    v4f a = *(const v4f*)(src + i);
    v4f b = *(const v4f*)(src + i + 4);
    v8bf o;
#pragma unroll
    for (int j = 0; j < 4; ++j) { o[j] = (__bf16)a[j]; o[j + 4] = (__bf16)b[j]; }
    *(v8bf*)(dst + i) = o;
}

// ---------------- projection GEMM: C[m][n] = sum_k A[m][k] * W[n][k] ---------
// A: f32 [ROWS][EMB] (in-register cvt), W: bf16 [EMB][EMB], C: bf16.
// Per-wave 16x64 tile.  TR_OUT=true writes C transposed as [EMB][ROWS] (for V)
// — packed 16B stores straight from the C-matrix VGPR layout.
template <bool TR_OUT>
__global__ void __launch_bounds__(256)
proj_gemm_kernel(const float* __restrict__ A, const __bf16* __restrict__ W,
                 __bf16* __restrict__ C) {
    const int tid = threadIdx.x;
    const int w = tid >> 5, lane = tid & 31, l = lane & 15, hf = lane >> 4;
    const int mBase = (blockIdx.x * 8 + w) * 16;
    const int nBase = blockIdx.y * 64;

    v8f acc[4] = {};
    for (int k0 = 0; k0 < EMB; k0 += 32) {
        v16bf a = load_a_f32(A, EMB, mBase, k0, l, hf);
#pragma unroll
        for (int j = 0; j < 4; ++j) {
            v16bf b = load_b_nt_bf16(W, EMB, nBase + j * 16, k0, l, hf);
            acc[j] = wmma_bf16(a, b, acc[j]);
        }
    }
    // C layout: vgpr r -> row M = r + hf*8, col N = l  (per 16x16 tile)
    if (TR_OUT) {
#pragma unroll
        for (int j = 0; j < 4; ++j) {
            v8bf pk;
#pragma unroll
            for (int r = 0; r < 8; ++r) pk[r] = (__bf16)acc[j][r];
            *(v8bf*)(C + (size_t)(nBase + j * 16 + l) * ROWS + mBase + hf * 8) = pk;
        }
    } else {
#pragma unroll
        for (int j = 0; j < 4; ++j)
#pragma unroll
            for (int r = 0; r < 8; ++r)
                C[(size_t)(mBase + r + hf * 8) * EMB + nBase + j * 16 + l] =
                    (__bf16)acc[j][r];
    }
}

// ---------------- fused attention with head-axis softmax ---------------------
// Block: 512 threads = 16 waves (one head each), one 16-query tile.
// The quirky softmax(dim=head) has NO reduction over k, so normalization is
// local to each (q,k) element once all 16 head scores are in LDS.
// LDS padding: scoresF h-stride 513 dwords (conflict-free h scans),
// attnB row stride 40 halves (16B-aligned ds_read_b128 A loads), h-stride 648.
#define SC_HS 513
#define AT_RS 40
#define AT_HS 648

__global__ void __launch_bounds__(512)
attn_kernel(const __bf16* __restrict__ Qp, const __bf16* __restrict__ Kp,
            const __bf16* __restrict__ Vt, __bf16* __restrict__ AO) {
    __shared__ float scoresF[HEADS * SC_HS];                  // 32.0 KB
    __shared__ __align__(16) __bf16 attnB[HEADS * AT_HS];     // 20.3 KB

    const int tid = threadIdx.x;
    const int h = tid >> 5, lane = tid & 31, l = lane & 15, hf = lane >> 4;
    const int qt = blockIdx.x, b = blockIdx.y;
    const int qBase = qt * 16;

    const __bf16* Qb = Qp + (size_t)b * SEQ * EMB;
    const __bf16* Kb = Kp + (size_t)b * SEQ * EMB;
    const __bf16* Vb = Vt + (size_t)b * SEQ;      // Vt is [EMB][ROWS]

    // Q A-operands for this head, kept in registers for the whole sweep.
    v16bf aq0 = load_a_bf16(Qb + h * HDIM, EMB, qBase, 0, l, hf);
    v16bf aq1 = load_a_bf16(Qb + h * HDIM, EMB, qBase, 32, l, hf);

    v8f o[4] = {};

    const int sq = tid >> 5;       // softmax element: q = tid/32, k = tid%32
    const int sk = tid & 31;

    for (int kt = 0; kt < SEQ; kt += 32) {
        // ---- scores: S_h = Q_h * K_h^T, two 16x16 tiles over 32 keys ----
#pragma unroll
        for (int kk = 0; kk < 32; kk += 16) {
            v8f sc = {};
            v16bf b0 = load_b_nt_bf16(Kb + h * HDIM, EMB, kt + kk, 0, l, hf);
            sc = wmma_bf16(aq0, b0, sc);
            v16bf b1 = load_b_nt_bf16(Kb + h * HDIM, EMB, kt + kk, 32, l, hf);
            sc = wmma_bf16(aq1, b1, sc);
#pragma unroll
            for (int r = 0; r < 8; ++r)
                scoresF[h * SC_HS + (r + hf * 8) * 32 + kk + l] = sc[r];
        }
        __syncthreads();

        // ---- softmax across the 16 heads for one (q,k) per thread ----
        {
            const int idx = sq * 32 + sk;
            float vals[HEADS];
            float m = -1e30f;
#pragma unroll
            for (int hh = 0; hh < HEADS; ++hh) {
                vals[hh] = scoresF[hh * SC_HS + idx] * 0.125f; // 1/sqrt(D)
                m = fmaxf(m, vals[hh]);
            }
            float sum = 0.f;
#pragma unroll
            for (int hh = 0; hh < HEADS; ++hh) {
                vals[hh] = __expf(vals[hh] - m);
                sum += vals[hh];
            }
            const float inv = 1.0f / sum;
#pragma unroll
            for (int hh = 0; hh < HEADS; ++hh)
                attnB[hh * AT_HS + sq * AT_RS + sk] = (__bf16)(vals[hh] * inv);
        }
        __syncthreads();

        // ---- AV: O_h += A_h(16x32) * V_h(32x64) ----
        v16bf aa;
        {
            const __bf16* ap = attnB + h * AT_HS + l * AT_RS + hf * 8;
            v8bf c0 = *(const v8bf*)ap;
            v8bf c1 = *(const v8bf*)(ap + 16);
#pragma unroll
            for (int i = 0; i < 8; ++i) { aa[i] = c0[i]; aa[i + 8] = c1[i]; }
        }
#pragma unroll
        for (int j = 0; j < 4; ++j) {
            v16bf bv = load_b_nt_bf16(Vb, ROWS, h * HDIM + j * 16, kt, l, hf);
            o[j] = wmma_bf16(aa, bv, o[j]);
        }
        // attnB is rewritten only after the next iteration's first barrier,
        // and scoresF only after this point in program order -> 2 barriers/iter.
    }

    // ---- store merged-head output (bf16) ----
#pragma unroll
    for (int j = 0; j < 4; ++j)
#pragma unroll
        for (int r = 0; r < 8; ++r)
            AO[(size_t)(b * SEQ + qBase + r + hf * 8) * EMB +
               h * HDIM + j * 16 + l] = (__bf16)o[j][r];
}

// ---------------- output projection: f32 out = AO(bf16) @ Wo^T(bf16) ---------
__global__ void __launch_bounds__(256)
out_gemm_kernel(const __bf16* __restrict__ A, const __bf16* __restrict__ W,
                float* __restrict__ C) {
    const int tid = threadIdx.x;
    const int w = tid >> 5, lane = tid & 31, l = lane & 15, hf = lane >> 4;
    const int mBase = (blockIdx.x * 8 + w) * 16;
    const int nBase = blockIdx.y * 64;

    v8f acc[4] = {};
    for (int k0 = 0; k0 < EMB; k0 += 32) {
        v16bf a = load_a_bf16(A, EMB, mBase, k0, l, hf);
#pragma unroll
        for (int j = 0; j < 4; ++j) {
            v16bf b = load_b_nt_bf16(W, EMB, nBase + j * 16, k0, l, hf);
            acc[j] = wmma_bf16(a, b, acc[j]);
        }
    }
#pragma unroll
    for (int j = 0; j < 4; ++j)
#pragma unroll
        for (int r = 0; r < 8; ++r)
            C[(size_t)(mBase + r + hf * 8) * EMB + nBase + j * 16 + l] =
                acc[j][r];
}

// ---------------- host launcher ----------------
extern "C" void kernel_launch(void* const* d_in, const int* in_sizes, int n_in,
                              void* d_out, int out_size, void* d_ws,
                              size_t ws_size, hipStream_t stream) {
    const float* q  = (const float*)d_in[0];
    const float* k  = (const float*)d_in[1];
    const float* v  = (const float*)d_in[2];
    const float* Wq = (const float*)d_in[3];
    const float* Wk = (const float*)d_in[4];
    const float* Wv = (const float*)d_in[5];
    const float* Wo = (const float*)d_in[6];

    // workspace layout (bf16):
    //   4 weight copies of EMB*EMB   (2 MB each)
    //   Qp, Kp, Vt, AO of ROWS*EMB   (8 MB each)   -> 40 MB total
    const size_t WN = (size_t)EMB * EMB;   // 1M
    const size_t N  = (size_t)ROWS * EMB;  // 4M
    __bf16* WqB = (__bf16*)d_ws;
    __bf16* WkB = WqB + WN;
    __bf16* WvB = WkB + WN;
    __bf16* WoB = WvB + WN;
    __bf16* Qp  = WoB + WN;
    __bf16* Kp  = Qp + N;
    __bf16* Vt  = Kp + N;        // transposed: [EMB][ROWS]
    __bf16* AO  = Vt + N;

    // one-time weight conversion: 1M elems / (256 thr * 8 elems) = 512 blocks
    const int cvtBlocks = (int)(WN / (256 * 8));
    cvt_f32_bf16_kernel<<<cvtBlocks, 256, 0, stream>>>(Wq, WqB);
    cvt_f32_bf16_kernel<<<cvtBlocks, 256, 0, stream>>>(Wk, WkB);
    cvt_f32_bf16_kernel<<<cvtBlocks, 256, 0, stream>>>(Wv, WvB);
    cvt_f32_bf16_kernel<<<cvtBlocks, 256, 0, stream>>>(Wo, WoB);

    dim3 gGemm(ROWS / (16 * 8), EMB / 64);  // (32, 16)
    proj_gemm_kernel<false><<<gGemm, 256, 0, stream>>>(q, WqB, Qp);
    proj_gemm_kernel<false><<<gGemm, 256, 0, stream>>>(k, WkB, Kp);
    proj_gemm_kernel<true ><<<gGemm, 256, 0, stream>>>(v, WvB, Vt);

    attn_kernel<<<dim3(SEQ / 16, BATCH), 512, 0, stream>>>(Qp, Kp, Vt, AO);

    out_gemm_kernel<<<gGemm, 256, 0, stream>>>(AO, WoB, (float*)d_out);
}
